// ODE_Attention_5677946765678
// MI455X (gfx1250) — matrix operand order repaired and checked by reading, not compiled, not run on hardware
//
#include <hip/hip_runtime.h>
#include <hip/hip_bf16.h>
#include <math.h>

// ---------------------------------------------------------------------------
// Problem constants (from reference): B,S,H,P,L,O,E = 32,1024,256,96,200,200,64
// ---------------------------------------------------------------------------
#define B_   32
#define S_   1024
#define H_   256
#define P_   96
#define L_   200
#define LP   224          // L and O padded to multiple of 32 for WMMA K-loop
#define E_   64
#define M_   (B_*S_)      // 32768 rows (b,s flattened)
#define N_EULER 4

typedef unsigned short u16;
typedef unsigned int   u32;
typedef __attribute__((ext_vector_type(16))) __bf16         v16bf;
typedef __attribute__((ext_vector_type(8)))  float          v8f;
typedef __attribute__((ext_vector_type(8)))  unsigned short v8u16;
typedef __attribute__((ext_vector_type(4)))  unsigned int   v4u;
typedef __attribute__((ext_vector_type(8)))  int            v8i;
typedef __attribute__((ext_vector_type(4)))  int            v4i;

struct Halves { v8u16 lo, hi; };
union  FragCast { Halves h; v16bf v; };

static __device__ __forceinline__ u16 f2bf(float x) {
  union { float f; unsigned u; } v; v.f = x;
  unsigned r = v.u + 0x7FFFu + ((v.u >> 16) & 1u);   // round-to-nearest-even
  return (u16)(r >> 16);
}
static __device__ __forceinline__ float bf2f(u16 x) {
  union { unsigned u; float f; } v; v.u = ((unsigned)x) << 16;
  return v.f;
}
static __device__ __forceinline__ v8f vzero8() {
  v8f c = {0.f,0.f,0.f,0.f,0.f,0.f,0.f,0.f};
  return c;
}

// A fragment (16x32 bf16) from row-major A[M][K].
static __device__ __forceinline__ v16bf load_a_frag(const u16* A, int lda,
                                                    int m0, int k0, int lane) {
  const int half = lane >> 4;
  const int m    = lane & 15;
  const u16* p = A + (size_t)(m0 + m) * lda + k0 + half * 8;
  FragCast f;
  f.h.lo = *(const v8u16*)(p);
  f.h.hi = *(const v8u16*)(p + 16);
  return f.v;
}
// B fragment (32x16 bf16) from N-major BT[N][K]: lane n = column, K in table order (8 at half*8, 8 at 16+half*8).
static __device__ __forceinline__ v16bf load_b_frag(const u16* BT, int ldb,
                                                    int n0, int k0, int lane) {
  const int half = lane >> 4;
  const int n    = lane & 15;
  const u16* p = BT + (size_t)(n0 + n) * ldb + k0 + half * 8;
  FragCast f;
  f.h.lo = *(const v8u16*)(p);
  f.h.hi = *(const v8u16*)(p + 16);
  return f.v;
}
static __device__ __forceinline__ v8f wmma_bf16(v16bf a, v16bf b, v8f c) {
  return __builtin_amdgcn_wmma_f32_16x16x32_bf16(false, a, false, b,
                                                 (short)0, c, false, false);
}

// ---------------------------------------------------------------------------
// Tensor Data Mover: 2D tile (width elems contiguous, height rows, row stride)
// of 2-byte elements, global -> LDS (packed per ISA D# group0/group1 layout).
// 6-arg builtin form (clang-23 / therock-10.0 headers).
// ---------------------------------------------------------------------------
static __device__ __forceinline__ void tdm_load_2d(unsigned lds_off,
                                                   const void* gptr,
                                                   unsigned width,
                                                   unsigned height,
                                                   unsigned stride) {
  unsigned long long ga = (unsigned long long)(uintptr_t)gptr;
  v4u g0 = { 1u,                                    // count=1 (valid user D#)
             lds_off,                               // lds_addr (bytes)
             (u32)(ga & 0xFFFFFFFFu),               // global_addr[31:0]
             (u32)((ga >> 32) & 0x01FFFFFFu) | (2u << 30) };   // addr[56:32], type=2
  v8i g1 = { (int)(1u << 16),                       // wg_mask=0, data_size=1 (2B)
             (int)((width & 0xFFFFu) << 16),        // tensor_dim0[15:0]
             (int)(((width >> 16) & 0xFFFFu) | ((height & 0xFFFFu) << 16)),
             (int)(((height >> 16) & 0xFFFFu) | ((width & 0xFFFFu) << 16)), // tile_dim0
             (int)(height & 0xFFFFu),               // tile_dim1 (tile_dim2 = 0)
             (int)stride,                           // tensor_dim0_stride[31:0]
             0, 0 };
  v4i gz4 = { 0, 0, 0, 0 };
  v8i gz8 = { 0, 0, 0, 0, 0, 0, 0, 0 };
  __builtin_amdgcn_tensor_load_to_lds(g0, g1, gz4, gz4, gz8, 0);
}

// ---------------------------------------------------------------------------
// Packing / small kernels
// ---------------------------------------------------------------------------
__global__ void k_f32_to_bf16(const float* __restrict__ src,
                              u16* __restrict__ dst, int n) {
  for (int i = blockIdx.x * blockDim.x + threadIdx.x; i < n;
       i += gridDim.x * blockDim.x)
    dst[i] = f2bf(src[i]);
}

// src fp32 [G][Ks][Ns] -> dst bf16 [G][Np][Kp] (output-major, zero padded)
__global__ void k_pack_T(const float* __restrict__ src, u16* __restrict__ dst,
                         int G, int Ks, int Ns, int Kp, int Np) {
  int total = G * Np * Kp;
  for (int i = blockIdx.x * blockDim.x + threadIdx.x; i < total;
       i += gridDim.x * blockDim.x) {
    int g = i / (Np * Kp);
    int r = i - g * (Np * Kp);
    int n = r / Kp;
    int k = r - n * Kp;
    float v = (n < Ns && k < Ks) ? src[((size_t)g * Ks + k) * Ns + n] : 0.f;
    dst[i] = f2bf(v);
  }
}

__global__ void k_eco(const float* __restrict__ eco, const float* __restrict__ W,
                      const float* __restrict__ bias, float* __restrict__ out) {
  int b = blockIdx.x, h = threadIdx.x;
  float acc = bias[h];
  for (int e = 0; e < E_; ++e) acc += eco[b * E_ + e] * W[e * H_ + h];
  out[b * H_ + h] = acc;
}

// ---------------------------------------------------------------------------
// WMMA GEMM core: one wave per block computes 64 rows x 32 cols (4x2 C tiles).
// B fragments amortized over 4 A fragments -> 1.5 b128 loads per WMMA.
// ---------------------------------------------------------------------------
static __device__ __forceinline__ void gemm_core64x32(const u16* A, int lda,
                                                      const u16* BT, int ldb,
                                                      int K, int m0, int n0,
                                                      int lane, v8f acc[4][2]) {
  for (int k0 = 0; k0 < K; k0 += 32) {
    v16bf b0 = load_b_frag(BT, ldb, n0,      k0, lane);
    v16bf b1 = load_b_frag(BT, ldb, n0 + 16, k0, lane);
#pragma unroll
    for (int mt = 0; mt < 4; ++mt) {
      v16bf a = load_a_frag(A, lda, m0 + mt * 16, k0, lane);
      acc[mt][0] = wmma_bf16(a, b0, acc[mt][0]);
      acc[mt][1] = wmma_bf16(a, b1, acc[mt][1]);
    }
  }
}
#define GEMM_INIT(acc)                                  \
  v8f acc[4][2];                                        \
  for (int mt = 0; mt < 4; ++mt) { acc[mt][0] = vzero8(); acc[mt][1] = vzero8(); }

__global__ void __launch_bounds__(32)
k_encode(const u16* __restrict__ Abf, const u16* __restrict__ WinT,
         const float* __restrict__ time_x, const float* __restrict__ Wt,
         const float* __restrict__ b_in, u16* __restrict__ hOut) {
  const int lane = threadIdx.x;
  const int n0 = blockIdx.x * 32, m0 = blockIdx.y * 64, k = blockIdx.z;
  const u16* BT = WinT + (size_t)k * LP * H_;
  GEMM_INIT(acc);
  gemm_core64x32(Abf, H_, BT, H_, H_, m0, n0, lane, acc);
  u16* hK = hOut + (size_t)k * M_ * LP;
  const int half = lane >> 4, ln = lane & 15;
  for (int mt = 0; mt < 4; ++mt)
    for (int t = 0; t < 2; ++t)
      for (int v = 0; v < 8; ++v) {
        int row = m0 + mt * 16 + v + 8 * half;
        int col = n0 + t * 16 + ln;
        float val = 0.f;
        if (col < L_) {
          val = acc[mt][t][v] + time_x[row] * Wt[k * L_ + col] + b_in[k * L_ + col];
          val = tanhf(val);
        }
        hK[(size_t)row * LP + col] = f2bf(val);
      }
}

__global__ void __launch_bounds__(32)
k_f1(const u16* __restrict__ hIn, const u16* __restrict__ WT,
     const float* __restrict__ bias, u16* __restrict__ hidOut) {
  const int lane = threadIdx.x;
  const int n0 = blockIdx.x * 32, m0 = blockIdx.y * 64, k = blockIdx.z;
  GEMM_INIT(acc);
  gemm_core64x32(hIn + (size_t)k * M_ * LP, LP, WT + (size_t)k * LP * LP, LP,
                 LP, m0, n0, lane, acc);
  u16* Ok = hidOut + (size_t)k * M_ * LP;
  const int half = lane >> 4, ln = lane & 15;
  for (int mt = 0; mt < 4; ++mt)
    for (int t = 0; t < 2; ++t)
      for (int v = 0; v < 8; ++v) {
        int row = m0 + mt * 16 + v + 8 * half;
        int col = n0 + t * 16 + ln;
        float val = (col < L_) ? tanhf(acc[mt][t][v] + bias[k * L_ + col]) : 0.f;
        Ok[(size_t)row * LP + col] = f2bf(val);
      }
}

__global__ void __launch_bounds__(32)
k_f2(const u16* __restrict__ hidIn, const u16* __restrict__ WT,
     const float* __restrict__ bias, const u16* __restrict__ hOld,
     u16* __restrict__ hNew) {
  const int lane = threadIdx.x;
  const int n0 = blockIdx.x * 32, m0 = blockIdx.y * 64, k = blockIdx.z;
  GEMM_INIT(acc);
  gemm_core64x32(hidIn + (size_t)k * M_ * LP, LP, WT + (size_t)k * LP * LP, LP,
                 LP, m0, n0, lane, acc);
  const u16* Ol = hOld + (size_t)k * M_ * LP;
  u16* Nw = hNew + (size_t)k * M_ * LP;
  const int half = lane >> 4, ln = lane & 15;
  for (int mt = 0; mt < 4; ++mt)
    for (int t = 0; t < 2; ++t)
      for (int v = 0; v < 8; ++v) {
        int row = m0 + mt * 16 + v + 8 * half;
        int col = n0 + t * 16 + ln;
        float val = 0.f;
        if (col < L_) {
          float f = acc[mt][t][v] + bias[k * L_ + col];
          val = bf2f(Ol[(size_t)row * LP + col]) + (1.0f / N_EULER) * f;
        }
        Nw[(size_t)row * LP + col] = f2bf(val);
      }
}

// QKV readout. Q,K stored row-major [seq][H]; V stored TRANSPOSED [b][h][s] so
// attention P.V B-fragments are contiguous along keys (no transpose staging).
__global__ void __launch_bounds__(32)
k_qkv(const u16* __restrict__ hIn, const u16* __restrict__ WT,
      const float* __restrict__ b_out, const float* __restrict__ eco_vec,
      const float* __restrict__ tran, u16* __restrict__ qkOut,
      u16* __restrict__ VtOut) {
  const int lane = threadIdx.x;
  const int n0 = blockIdx.x * 32, m0 = blockIdx.y * 64, k = blockIdx.z;
  GEMM_INIT(acc);
  gemm_core64x32(hIn + (size_t)k * M_ * LP, LP, WT + (size_t)k * H_ * LP, LP,
                 LP, m0, n0, lane, acc);
  const int half = lane >> 4, ln = lane & 15;
  for (int mt = 0; mt < 4; ++mt)
    for (int t = 0; t < 2; ++t)
      for (int v = 0; v < 8; ++v) {
        int row = m0 + mt * 16 + v + 8 * half;
        int col = n0 + t * 16 + ln;
        int bb  = row >> 10;
        float val = acc[mt][t][v] + b_out[k * H_ + col] + eco_vec[bb * H_ + col];
        if (tran[(size_t)row * H_ + (H_ - 1)] == 0.f) val = 0.f;   // pad mask
        u16 bv = f2bf(val);
        if (k == 2) VtOut[((size_t)bb * H_ + col) * S_ + (row & (S_ - 1))] = bv;
        else        qkOut[((size_t)k * M_ + row) * H_ + col] = bv;
      }
}

__global__ void __launch_bounds__(32)
k_lin(const u16* __restrict__ ctx, const u16* __restrict__ WT,
      const float* __restrict__ b_lin, float* __restrict__ out) {
  const int lane = threadIdx.x;
  const int n0 = blockIdx.x * 32, m0 = blockIdx.y * 64;
  GEMM_INIT(acc);
  gemm_core64x32(ctx, H_, WT, H_, H_, m0, n0, lane, acc);
  const int half = lane >> 4, ln = lane & 15;
  for (int mt = 0; mt < 4; ++mt)
    for (int t = 0; t < 2; ++t)
      for (int v = 0; v < 8; ++v) {
        int row = m0 + mt * 16 + v + 8 * half;
        int col = n0 + t * 16 + ln;
        out[(size_t)row * P_ + col] = acc[mt][t][v] + b_lin[col];
      }
}

// ---------------------------------------------------------------------------
// Flash attention: 256 threads = 8 waves, each wave owns a 16-query tile with
// full-H fp32 accumulators. Per 32-key chunk, wave 0 stages the K chunk
// (32x256 row-major) and the V^T chunk (256x32) into LDS via the Tensor Data
// Mover, waits TENSORcnt, and a barrier publishes them to all 8 waves.
// Q.K^T B-fragments and P.V B-fragments are then contiguous LDS b128 reads.
// ---------------------------------------------------------------------------
__global__ void __launch_bounds__(256)
k_attn(const u16* __restrict__ qk, const u16* __restrict__ VtG,
       u16* __restrict__ ctxOut) {
  __shared__ __align__(16) u16 klds[32 * H_];        // [key][h]   16 KB
  __shared__ __align__(16) u16 vlds[H_ * 32];        // [h][key]   16 KB
  __shared__ __align__(16) u16 pbuf[8][16 * 32];     // per-wave P  8 KB

  const int tid  = threadIdx.x;
  const int lane = tid & 31;
  const int wave = tid >> 5;
  const int b    = blockIdx.y;
  const int q0   = blockIdx.x * 128 + wave * 16;

  const u16* Qg = qk + ((size_t)0 * M_ + (size_t)b * S_ + q0) * H_;
  const u16* Kg = qk + ((size_t)1 * M_ + (size_t)b * S_) * H_;
  const u16* Vt = VtG + (size_t)b * H_ * S_;

  const unsigned klds_off = (unsigned)(uintptr_t)(&klds[0]);
  const unsigned vlds_off = (unsigned)(uintptr_t)(&vlds[0]);

  v16bf qa[8];                                   // Q tile 16 x 256 in VGPRs
  for (int kk = 0; kk < 8; ++kk) qa[kk] = load_a_frag(Qg, H_, 0, kk * 32, lane);

  v8f acc[16];                                   // ctx 16 x 256 fp32
  for (int t = 0; t < 16; ++t) acc[t] = vzero8();
  float m_i[8], l_i[8];
  for (int v = 0; v < 8; ++v) { m_i[v] = -1e30f; l_i[v] = 0.f; }

  const int half = lane >> 4, ln = lane & 15;
  const float scale = 0.0625f;                   // 1/sqrt(256)

  for (int kt = 0; kt < S_; kt += 32) {
    if (wave == 0) {                             // TDM stage K + V^T chunks
      tdm_load_2d(klds_off, Kg + (size_t)kt * H_, H_, 32, H_);
      tdm_load_2d(vlds_off, Vt + kt, 32, H_, S_);
      __builtin_amdgcn_s_wait_tensorcnt(0);
    }
    __syncthreads();

    if (kt + 32 < S_) {                          // prefetch next chunk into L2
      __builtin_prefetch(Kg + (size_t)(kt + 32) * H_ + (size_t)lane * 256, 0, 1);
      __builtin_prefetch(Vt + (size_t)lane * S_ * 8 + kt + 32, 0, 1);
    }

    // scores S = Q . K^T for 32 keys (2 C tiles), contraction over H=256
    v8f s0 = vzero8(), s1 = vzero8();
    for (int kk = 0; kk < 8; ++kk) {
      v16bf b0 = load_b_frag(klds, H_, 0,  kk * 32, lane);
      v16bf b1 = load_b_frag(klds, H_, 16, kk * 32, lane);
      s0 = wmma_bf16(qa[kk], b0, s0);
      s1 = wmma_bf16(qa[kk], b1, s1);
    }

    // online softmax per row (row = v + 8*half, over 16 lanes of a half)
    for (int v = 0; v < 8; ++v) {
      float a0 = s0[v] * scale, a1 = s1[v] * scale;
      float mx = fmaxf(a0, a1);
      for (int off = 1; off < 16; off <<= 1)
        mx = fmaxf(mx, __shfl_xor(mx, off, 32));
      float mnew = fmaxf(m_i[v], mx);
      float fac  = __expf(m_i[v] - mnew);
      float p0 = __expf(a0 - mnew);
      float p1 = __expf(a1 - mnew);
      float rs = p0 + p1;
      for (int off = 1; off < 16; off <<= 1)
        rs += __shfl_xor(rs, off, 32);
      l_i[v] = l_i[v] * fac + rs;
      m_i[v] = mnew;
      for (int t = 0; t < 16; ++t) acc[t][v] *= fac;
      int prow = v + 8 * half;                   // C-layout -> row-major P in LDS
      pbuf[wave][prow * 32 + ln]      = f2bf(p0);
      pbuf[wave][prow * 32 + 16 + ln] = f2bf(p1);
    }
    asm volatile("s_wait_dscnt 0" ::: "memory"); // P store -> A-frag load
    v16bf pa = load_a_frag(&pbuf[wave][0], 32, 0, 0, lane);
#pragma unroll
    for (int t = 0; t < 16; ++t) {               // ctx += P . V
      v16bf bv = load_b_frag(vlds, 32, t * 16, 0, lane);
      acc[t] = wmma_bf16(pa, bv, acc[t]);
    }
    __syncthreads();                             // all reads done before next TDM
  }

  // normalize and write ctx (bf16)
  u16* co = ctxOut + ((size_t)b * S_ + q0) * H_;
  for (int v = 0; v < 8; ++v) {
    float l   = l_i[v];
    float inv = (l > 0.f) ? 1.f / l : 0.f;
    int row = v + 8 * half;
    for (int t = 0; t < 16; ++t) {
      int col = t * 16 + ln;
      co[(size_t)row * H_ + col] = f2bf(acc[t][v] * inv);
    }
  }
}

// ---------------------------------------------------------------------------
// Host launcher
// ---------------------------------------------------------------------------
extern "C" void kernel_launch(void* const* d_in, const int* in_sizes, int n_in,
                              void* d_out, int out_size, void* d_ws, size_t ws_size,
                              hipStream_t stream) {
  (void)in_sizes; (void)n_in; (void)out_size; (void)ws_size;
  const float* eco_data = (const float*)d_in[0];
  const float* tran     = (const float*)d_in[1];
  const float* time_x   = (const float*)d_in[3];   // concat+slice == time_x
  const float* W_eco    = (const float*)d_in[6];
  const float* b_eco    = (const float*)d_in[7];
  const float* W_in     = (const float*)d_in[8];
  const float* W_t      = (const float*)d_in[9];
  const float* b_in     = (const float*)d_in[10];
  const float* W_f1     = (const float*)d_in[11];
  const float* b_f1     = (const float*)d_in[12];
  const float* W_f2     = (const float*)d_in[13];
  const float* b_f2     = (const float*)d_in[14];
  const float* W_out    = (const float*)d_in[15];
  const float* b_out    = (const float*)d_in[16];
  const float* W_lin    = (const float*)d_in[17];
  const float* b_lin    = (const float*)d_in[18];
  float* out = (float*)d_out;

  char* ws = (char*)d_ws;
  size_t off = 0;
  auto alloc = [&](size_t bytes) -> void* {
    void* p = ws + off;
    off = (off + bytes + 255) & ~(size_t)255;
    return p;
  };
  u16*   tran_bf = (u16*)alloc((size_t)M_ * H_ * 2);
  u16*   hA      = (u16*)alloc((size_t)3 * M_ * LP * 2);
  u16*   hB      = (u16*)alloc((size_t)3 * M_ * LP * 2);
  u16*   hid     = (u16*)alloc((size_t)3 * M_ * LP * 2);
  u16*   qk      = (u16*)alloc((size_t)2 * M_ * H_ * 2);   // Q,K row-major
  u16*   Vt      = (u16*)alloc((size_t)M_ * H_ * 2);       // V transposed [b][h][s]
  u16*   ctx     = (u16*)alloc((size_t)M_ * H_ * 2);
  float* eco_vec = (float*)alloc((size_t)B_ * H_ * 4);
  u16*   WinT    = (u16*)alloc((size_t)3 * LP * H_ * 2);
  u16*   Wf1T    = (u16*)alloc((size_t)3 * LP * LP * 2);
  u16*   Wf2T    = (u16*)alloc((size_t)3 * LP * LP * 2);
  u16*   WoutT   = (u16*)alloc((size_t)3 * H_ * LP * 2);
  u16*   WlinT   = (u16*)alloc((size_t)P_ * H_ * 2);

  k_f32_to_bf16<<<dim3(2048), dim3(256), 0, stream>>>(tran, tran_bf, M_ * H_);
  k_pack_T<<<dim3(512), dim3(256), 0, stream>>>(W_in,  WinT,  3, H_, L_, H_, LP);
  k_pack_T<<<dim3(512), dim3(256), 0, stream>>>(W_f1,  Wf1T,  3, L_, L_, LP, LP);
  k_pack_T<<<dim3(512), dim3(256), 0, stream>>>(W_f2,  Wf2T,  3, L_, L_, LP, LP);
  k_pack_T<<<dim3(512), dim3(256), 0, stream>>>(W_out, WoutT, 3, L_, H_, LP, H_);
  k_pack_T<<<dim3(128), dim3(256), 0, stream>>>(W_lin, WlinT, 1, H_, P_, H_, P_);

  k_eco<<<dim3(B_), dim3(H_), 0, stream>>>(eco_data, W_eco, b_eco, eco_vec);

  k_encode<<<dim3(LP / 32, M_ / 64, 3), dim3(32), 0, stream>>>(
      tran_bf, WinT, time_x, W_t, b_in, hA);

  u16* hX = hA; u16* hY = hB;
  for (int it = 0; it < N_EULER; ++it) {
    k_f1<<<dim3(LP / 32, M_ / 64, 3), dim3(32), 0, stream>>>(hX, Wf1T, b_f1, hid);
    k_f2<<<dim3(LP / 32, M_ / 64, 3), dim3(32), 0, stream>>>(hid, Wf2T, b_f2, hX, hY);
    u16* t = hX; hX = hY; hY = t;
  }

  k_qkv<<<dim3(H_ / 32, M_ / 64, 3), dim3(32), 0, stream>>>(
      hX, WoutT, b_out, eco_vec, tran, qk, Vt);

  k_attn<<<dim3(S_ / 128, B_), dim3(256), 0, stream>>>(qk, Vt, ctx);

  k_lin<<<dim3(P_ / 32, M_ / 64, 1), dim3(32), 0, stream>>>(ctx, WlinT, b_lin, out);
}
